// T5_34900904248069
// MI455X (gfx1250) — matrix operand order repaired and checked by reading, not compiled, not run on hardware
//
#include <hip/hip_runtime.h>
#include <hip/hip_bf16.h>
#include <math.h>

typedef _Float16 v16h __attribute__((ext_vector_type(16)));
typedef _Float16 v8h  __attribute__((ext_vector_type(8)));
typedef _Float16 v4h  __attribute__((ext_vector_type(4)));
typedef float    v8f  __attribute__((ext_vector_type(8)));

#define WMMA_F16(a, b, c) \
  __builtin_amdgcn_wmma_f32_16x16x32_f16(false, (a), false, (b), (short)0, (c), false, false)

static __device__ __forceinline__ v16h cat8(v8h lo, v8h hi) {
  return __builtin_shufflevector(lo, hi, 0, 1, 2, 3, 4, 5, 6, 7, 8, 9, 10, 11, 12, 13, 14, 15);
}

static __device__ __forceinline__ v4h pack4(float4 f) {
  v4h r;
  r[0] = (_Float16)f.x; r[1] = (_Float16)f.y; r[2] = (_Float16)f.z; r[3] = (_Float16)f.w;
  return r;
}

// ---------------------------------------------------------------------------
// GEMM: D[M,N] = A[M,K] @ B[K,N] (+ R), all f32 row-major, f16 WMMA, f32 accum.
// Block tile 128x128, K-step 32, software-pipelined global->LDS staging.
// 8 waves: 4(M) x 2(N); wave tile 32x64. Requires M%128==0, N%128==0, K%32==0.
// ---------------------------------------------------------------------------
template <bool HAS_R>
__global__ __launch_bounds__(256) void gemm_f16_kernel(
    const float* __restrict__ A, const float* __restrict__ B,
    const float* R, float* D, int M, int N, int K) {
  __shared__ _Float16 As[128][48];  // [m][k], row stride 96B (16B aligned)
  __shared__ _Float16 Bs[128][48];  // [n][k] (transposed)
  const int tid = threadIdx.x;
  const int lane = tid & 31, wave = tid >> 5;
  const int wm = wave & 3, wn = wave >> 2;
  const int lmod = lane & 15, lhalf = lane >> 4;
  const int bm = blockIdx.y * 128, bn = blockIdx.x * 128;

  v8f acc[2][4] = {};
  float4 ar[4], br[4];

  auto load_tile = [&](int k0) {
#pragma unroll
    for (int it = 0; it < 4; ++it) {
      int idx = tid + it * 256;
      int r = idx >> 3, c = (idx & 7) * 4;
      ar[it] = *(const float4*)(A + (size_t)(bm + r) * K + k0 + c);
      int rb = idx >> 5, cb = (idx & 31) * 4;
      br[it] = *(const float4*)(B + (size_t)(k0 + rb) * N + bn + cb);
    }
  };
  auto store_tile = [&]() {
#pragma unroll
    for (int it = 0; it < 4; ++it) {
      int idx = tid + it * 256;
      int r = idx >> 3, c = (idx & 7) * 4;
      *(v4h*)&As[r][c] = pack4(ar[it]);  // one ds_store_b64
      int rb = idx >> 5, cb = (idx & 31) * 4;
      Bs[cb + 0][rb] = (_Float16)br[it].x;
      Bs[cb + 1][rb] = (_Float16)br[it].y;
      Bs[cb + 2][rb] = (_Float16)br[it].z;
      Bs[cb + 3][rb] = (_Float16)br[it].w;
    }
  };

  load_tile(0);
  for (int k0 = 0; k0 < K; k0 += 32) {
    store_tile();
    __syncthreads();
    if (k0 + 32 < K) load_tile(k0 + 32);  // prefetch next tile during WMMAs

    // A-frag: lane m=lmod; halves0-7: K=lhalf*8+h; halves8-15: K=16+lhalf*8+h
    v16h af[2], bf[4];
    const int akb = lhalf * 8;
#pragma unroll
    for (int i = 0; i < 2; ++i) {
      int m = wm * 32 + i * 16 + lmod;
      af[i] = cat8(*(const v8h*)&As[m][akb], *(const v8h*)&As[m][16 + akb]);
    }
    // B-frag: lane n=lmod; halves0-7: K=lhalf*8+h; halves8-15: K=16+lhalf*8+h
    const int bkb = lhalf * 8;
#pragma unroll
    for (int j = 0; j < 4; ++j) {
      int n = wn * 64 + j * 16 + lmod;
      bf[j] = cat8(*(const v8h*)&Bs[n][bkb], *(const v8h*)&Bs[n][16 + bkb]);
    }
#pragma unroll
    for (int i = 0; i < 2; ++i)
#pragma unroll
      for (int j = 0; j < 4; ++j)
        acc[i][j] = WMMA_F16(af[i], bf[j], acc[i][j]);
    __syncthreads();
  }

  // Epilogue: C/D layout lane n=lmod; vgpr v -> row m = v + 8*lhalf.
  const size_t base = (size_t)(bm + wm * 32 + 8 * lhalf) * N + (bn + wn * 64 + lmod);
  float* Dp = D + base;
#pragma unroll
  for (int i = 0; i < 2; ++i) {
#pragma unroll
    for (int v = 0; v < 8; ++v) {
      const size_t ro = (size_t)(i * 16 + v) * N;
      float r0 = 0.f, r1 = 0.f, r2 = 0.f, r3 = 0.f;
      if (HAS_R) {
        const float* Rp = R + base + ro;
        r0 = Rp[0]; r1 = Rp[16]; r2 = Rp[32]; r3 = Rp[48];
      }
      Dp[ro + 0]  = acc[i][0][v] + r0;
      Dp[ro + 16] = acc[i][1][v] + r1;
      Dp[ro + 32] = acc[i][2][v] + r2;
      Dp[ro + 48] = acc[i][3][v] + r3;
    }
  }
}

// ---------------------------------------------------------------------------
// Flash attention, WMMA everywhere. 1 wave = 16 queries; 32 keys / iteration.
// scores = (Q.K^T + bias[h][q-k]) * sqrt(64); optional causal mask.
// Grid: (Tq/64, H, B); block 128 (4 waves).
// ---------------------------------------------------------------------------
__global__ __launch_bounds__(128) void attn_kernel(
    const float* __restrict__ Qp, int sq, const float* __restrict__ Kp, int sk,
    const float* __restrict__ Vp, int sv, float* __restrict__ Op, int so,
    const float* __restrict__ btab, int Tq, int Tk, int causal) {
  __shared__ _Float16 Pl[4][16][40];  // per-wave 16x32 P tile, stride 80B
  const int lane = threadIdx.x & 31, wave = threadIdx.x >> 5;
  const int lmod = lane & 15, lhalf = lane >> 4;
  const int h = blockIdx.y, b = blockIdx.z;
  const int qtile = blockIdx.x * 64 + wave * 16;
  const float* bt = btab + h * 2047;
  const float* qb = Qp + (size_t)b * Tq * sq + h * 64;
  const float* kb = Kp + (size_t)b * Tk * sk + h * 64;
  const float* vb = Vp + (size_t)b * Tk * sv + h * 64;

  // Q fragments (A-matrix 16x32, two d-chunks)
  v16h qa[2];
  {
    const float* qrow = qb + (size_t)(qtile + lmod) * sq;
#pragma unroll
    for (int c = 0; c < 2; ++c) {
      int d0 = c * 32 + lhalf * 8;
      float4 f0 = *(const float4*)(qrow + d0);
      float4 f1 = *(const float4*)(qrow + d0 + 4);
      float4 f2 = *(const float4*)(qrow + d0 + 16);
      float4 f3 = *(const float4*)(qrow + d0 + 20);
      v16h a;
      a[0] = (_Float16)f0.x; a[1] = (_Float16)f0.y; a[2] = (_Float16)f0.z; a[3] = (_Float16)f0.w;
      a[4] = (_Float16)f1.x; a[5] = (_Float16)f1.y; a[6] = (_Float16)f1.z; a[7] = (_Float16)f1.w;
      a[8] = (_Float16)f2.x; a[9] = (_Float16)f2.y; a[10] = (_Float16)f2.z; a[11] = (_Float16)f2.w;
      a[12] = (_Float16)f3.x; a[13] = (_Float16)f3.y; a[14] = (_Float16)f3.z; a[15] = (_Float16)f3.w;
      qa[c] = a;
    }
  }
  v8f o[4] = {};
  float mrow[8], lrow[8];
#pragma unroll
  for (int v = 0; v < 8; ++v) { mrow[v] = -1e30f; lrow[v] = 0.0f; }

  const int niter = causal ? (qtile / 32 + 1) : (Tk / 32);
  for (int it = 0; it < niter; ++it) {
    const int kt = it * 32;
    // S = Q K^T for 2 key tiles of 16
    v8f s[2] = {};
#pragma unroll
    for (int j = 0; j < 2; ++j) {
      const float* krow = kb + (size_t)(kt + j * 16 + lmod) * sk;
#pragma unroll
      for (int c = 0; c < 2; ++c) {
        int d0 = c * 32 + lhalf * 8;
        float4 f0 = *(const float4*)(krow + d0);
        float4 f1 = *(const float4*)(krow + d0 + 4);
        float4 f2 = *(const float4*)(krow + d0 + 16);
        float4 f3 = *(const float4*)(krow + d0 + 20);
        v16h kf;
        kf[0] = (_Float16)f0.x; kf[1] = (_Float16)f0.y; kf[2] = (_Float16)f0.z; kf[3] = (_Float16)f0.w;
        kf[4] = (_Float16)f1.x; kf[5] = (_Float16)f1.y; kf[6] = (_Float16)f1.z; kf[7] = (_Float16)f1.w;
        kf[8] = (_Float16)f2.x; kf[9] = (_Float16)f2.y; kf[10] = (_Float16)f2.z; kf[11] = (_Float16)f2.w;
        kf[12] = (_Float16)f3.x; kf[13] = (_Float16)f3.y; kf[14] = (_Float16)f3.z; kf[15] = (_Float16)f3.w;
        s[j] = WMMA_F16(qa[c], kf, s[j]);
      }
    }
    // bias + scale + mask
#pragma unroll
    for (int j = 0; j < 2; ++j) {
      int kg = kt + j * 16 + lmod;
#pragma unroll
      for (int v = 0; v < 8; ++v) {
        int qg = qtile + v + 8 * lhalf;
        float sc = (s[j][v] + bt[qg - kg + 1023]) * 8.0f;
        if (causal && kg > qg) sc = -1e30f;
        s[j][v] = sc;
      }
    }
    // online softmax (row reductions across 16-lane halves)
#pragma unroll
    for (int v = 0; v < 8; ++v) {
      float rmax = fmaxf(s[0][v], s[1][v]);
#pragma unroll
      for (int off = 8; off > 0; off >>= 1) rmax = fmaxf(rmax, __shfl_xor(rmax, off, 32));
      float mnew = fmaxf(mrow[v], rmax);
      float p0 = __expf(s[0][v] - mnew);
      float p1 = __expf(s[1][v] - mnew);
      float rsum = p0 + p1;
#pragma unroll
      for (int off = 8; off > 0; off >>= 1) rsum += __shfl_xor(rsum, off, 32);
      float scold = __expf(mrow[v] - mnew);
      lrow[v] = lrow[v] * scold + rsum;
      mrow[v] = mnew;
#pragma unroll
      for (int dj = 0; dj < 4; ++dj) o[dj][v] *= scold;
      s[0][v] = p0;
      s[1][v] = p1;
    }
    // P (C-layout) -> LDS -> A-fragment layout
#pragma unroll
    for (int v = 0; v < 8; ++v) {
      int m = v + 8 * lhalf;
      Pl[wave][m][lmod] = (_Float16)s[0][v];
      Pl[wave][m][16 + lmod] = (_Float16)s[1][v];
    }
    asm volatile("s_wait_dscnt 0" ::: "memory");  // cross-lane LDS visibility in-wave
    v16h pa;
    {
      const _Float16* prow = &Pl[wave][lmod][0];
      int kb8 = lhalf * 8;
      pa = cat8(*(const v8h*)(prow + kb8), *(const v8h*)(prow + 16 + kb8));
    }
    // O += P @ V  (V B-frags: n = d col, K = key)
#pragma unroll
    for (int dj = 0; dj < 4; ++dj) {
      int dcol = dj * 16 + lmod;
      v16h vf;
#pragma unroll
      for (int i = 0; i < 16; ++i)
        vf[i] = (_Float16)vb[(size_t)(kt + lhalf * 8 + (i & 7) + ((i >> 3) << 4)) * sv + dcol];
      o[dj] = WMMA_F16(pa, vf, o[dj]);
    }
  }
  // normalize + store merged-head output
#pragma unroll
  for (int v = 0; v < 8; ++v) {
    float inv = 1.0f / lrow[v];
    int m = qtile + v + 8 * lhalf;
    float* orow = Op + (size_t)(b * Tq + m) * so + h * 64;
#pragma unroll
    for (int dj = 0; dj < 4; ++dj) orow[dj * 16 + lmod] = o[dj][v] * inv;
  }
}

// ---------------------------------------------------------------------------
// LayerNorm (weight only, eps 1e-5). One 128-thread block per 512-wide row.
// ---------------------------------------------------------------------------
__global__ __launch_bounds__(128) void ln_kernel(const float* __restrict__ X,
                                                 const float* __restrict__ W,
                                                 float* __restrict__ Y) {
  __shared__ float ps[4], ps2[4];
  const int row = blockIdx.x, t = threadIdx.x;
  const float4 xv = *(const float4*)(X + (size_t)row * 512 + t * 4);
  float s = xv.x + xv.y + xv.z + xv.w;
  float s2 = xv.x * xv.x + xv.y * xv.y + xv.z * xv.z + xv.w * xv.w;
#pragma unroll
  for (int off = 16; off > 0; off >>= 1) {
    s += __shfl_xor(s, off, 32);
    s2 += __shfl_xor(s2, off, 32);
  }
  if ((t & 31) == 0) { ps[t >> 5] = s; ps2[t >> 5] = s2; }
  __syncthreads();
  s = ps[0] + ps[1] + ps[2] + ps[3];
  s2 = ps2[0] + ps2[1] + ps2[2] + ps2[3];
  const float mean = s * (1.0f / 512.0f);
  const float var = s2 * (1.0f / 512.0f) - mean * mean;
  const float inv = rsqrtf(var + 1e-5f);
  const float4 wv = *(const float4*)(W + t * 4);
  float4 yv;
  yv.x = (xv.x - mean) * inv * wv.x;
  yv.y = (xv.y - mean) * inv * wv.y;
  yv.z = (xv.z - mean) * inv * wv.z;
  yv.w = (xv.w - mean) * inv * wv.w;
  *(float4*)(Y + (size_t)row * 512 + t * 4) = yv;
}

__global__ __launch_bounds__(128) void embed_kernel(const int* __restrict__ idx,
                                                    const float* __restrict__ wte,
                                                    float* __restrict__ out) {
  const int row = blockIdx.x;
  const int tok = idx[row];
  const float4 v = *(const float4*)(wte + (size_t)tok * 512 + threadIdx.x * 4);
  *(float4*)(out + (size_t)row * 512 + threadIdx.x * 4) = v;
}

__global__ __launch_bounds__(256) void geglu_kernel(const float* __restrict__ G,
                                                    const float* __restrict__ U,
                                                    float* __restrict__ Out, int n) {
  int i = blockIdx.x * 256 + threadIdx.x;
  if (i < n) {
    float x = G[i];
    float t = 0.7978845608028654f * (x + 0.044715f * x * x * x);
    Out[i] = 0.5f * x * (1.0f + tanhf(t)) * U[i];
  }
}

// T5 relative-position bias tables: tab[variant][h][delta+1023], delta = q-k.
// variant 0 = bidirectional (enc self + cross), 1 = unidirectional (dec self).
__global__ __launch_bounds__(256) void biastab_kernel(const float* __restrict__ wpe,
                                                      float* __restrict__ tab) {
  int i = blockIdx.x * 256 + threadIdx.x;
  if (i >= 2 * 8 * 2047) return;
  int variant = i / (8 * 2047);
  int rem = i - variant * 8 * 2047;
  int h = rem / 2047;
  int rp = (rem % 2047) - 1023;  // rp = q - k
  int bucket;
  if (variant == 0) {
    int res = 0;
    if (rp < 0) { res = 16; rp = -rp; }
    if (rp < 8) bucket = res + rp;
    else {
      int v = 8 + (int)(logf((float)rp / 8.0f) / logf(16.0f) * 8.0f);
      bucket = res + (v < 15 ? v : 15);
    }
  } else {
    if (rp < 0) rp = 0;
    if (rp < 16) bucket = rp;
    else {
      int v = 16 + (int)(logf((float)rp / 16.0f) / logf(8.0f) * 16.0f);
      bucket = (v < 31 ? v : 31);
    }
  }
  tab[i] = wpe[bucket * 8 + h];
}

__global__ __launch_bounds__(256) void transpose_kernel(const float* __restrict__ wte,
                                                        float* __restrict__ wteT) {
  int i = blockIdx.x * 256 + threadIdx.x;
  if (i < 384 * 512) {
    int v = i / 512, c = i - v * 512;
    wteT[(size_t)c * 384 + v] = wte[i];
  }
}

// ---------------------------------------------------------------------------
static inline void gemm(const float* A, const float* B, const float* R, float* D,
                        int M, int N, int K, hipStream_t s) {
  dim3 g(N / 128, M / 128);
  if (R)
    gemm_f16_kernel<true><<<g, 256, 0, s>>>(A, B, R, D, M, N, K);
  else
    gemm_f16_kernel<false><<<g, 256, 0, s>>>(A, B, R, D, M, N, K);
}

extern "C" void kernel_launch(void* const* d_in, const int* in_sizes, int n_in,
                              void* d_out, int out_size, void* d_ws, size_t ws_size,
                              hipStream_t stream) {
  (void)in_sizes; (void)n_in; (void)out_size; (void)ws_size;
  const int* src_idx = (const int*)d_in[0];
  const int* trg_idx = (const int*)d_in[1];
  const float* wte = (const float*)d_in[2];
  const float* rel_wpe = (const float*)d_in[3];
  const float* enc_ln1 = (const float*)d_in[4];
  const float* enc_wqkv = (const float*)d_in[5];
  const float* enc_wproj = (const float*)d_in[6];
  const float* enc_ln2 = (const float*)d_in[7];
  const float* enc_wgate = (const float*)d_in[8];
  const float* enc_wup = (const float*)d_in[9];
  const float* enc_wdown = (const float*)d_in[10];
  const float* ln_enc = (const float*)d_in[11];
  const float* dec_ln1 = (const float*)d_in[12];
  const float* dec_wqkv = (const float*)d_in[13];
  const float* dec_wproj = (const float*)d_in[14];
  const float* dec_ln2 = (const float*)d_in[15];
  const float* dec_wq = (const float*)d_in[16];
  const float* dec_wkv = (const float*)d_in[17];
  const float* dec_wcproj = (const float*)d_in[18];
  const float* dec_ln3 = (const float*)d_in[19];
  const float* dec_wgate = (const float*)d_in[20];
  const float* dec_wup = (const float*)d_in[21];
  const float* dec_wdown = (const float*)d_in[22];
  const float* ln_dec = (const float*)d_in[23];
  float* out = (float*)d_out;

  // Workspace layout (floats); total ~29.6M floats (~118 MB).
  float* ws = (float*)d_ws;
  float* buf_x = ws;                     // 4096*512
  float* buf_enc = buf_x + 2097152;      // 4096*512
  float* buf_ln = buf_enc + 2097152;     // 4096*512
  float* buf_attn = buf_ln + 2097152;    // 4096*512
  float* buf_kv = buf_attn + 2097152;    // 4096*1024
  float* ff1 = buf_kv + 4194304;         // 4096*2048 (also qkv)
  float* ff2 = ff1 + 8388608;            // 4096*2048 (also cross-q)
  float* tab = ff2 + 8388608;            // 2*8*2047
  float* wteT = tab + 32768;             // 512*384

  const float* tab_bidir = tab;
  const float* tab_caus = tab + 8 * 2047;

  biastab_kernel<<<(2 * 8 * 2047 + 255) / 256, 256, 0, stream>>>(rel_wpe, tab);
  transpose_kernel<<<(384 * 512 + 255) / 256, 256, 0, stream>>>(wte, wteT);

  // ---------------- encoder ----------------
  embed_kernel<<<4096, 128, 0, stream>>>(src_idx, wte, buf_x);
  for (int i = 0; i < 4; ++i) {
    ln_kernel<<<4096, 128, 0, stream>>>(buf_x, enc_ln1 + i * 512, buf_ln);
    gemm(buf_ln, enc_wqkv + (size_t)i * 512 * 1536, nullptr, ff1, 4096, 1536, 512, stream);
    attn_kernel<<<dim3(16, 8, 4), 128, 0, stream>>>(ff1, 1536, ff1 + 512, 1536, ff1 + 1024, 1536,
                                                    buf_attn, 512, tab_bidir, 1024, 1024, 0);
    gemm(buf_attn, enc_wproj + (size_t)i * 512 * 512, buf_x, buf_x, 4096, 512, 512, stream);
    ln_kernel<<<4096, 128, 0, stream>>>(buf_x, enc_ln2 + i * 512, buf_ln);
    gemm(buf_ln, enc_wgate + (size_t)i * 512 * 2048, nullptr, ff1, 4096, 2048, 512, stream);
    gemm(buf_ln, enc_wup + (size_t)i * 512 * 2048, nullptr, ff2, 4096, 2048, 512, stream);
    geglu_kernel<<<(8388608 + 255) / 256, 256, 0, stream>>>(ff1, ff2, ff1, 8388608);
    gemm(ff1, enc_wdown + (size_t)i * 2048 * 512, buf_x, buf_x, 4096, 512, 2048, stream);
  }
  ln_kernel<<<4096, 128, 0, stream>>>(buf_x, ln_enc, buf_enc);

  // ---------------- decoder ----------------
  embed_kernel<<<4096, 128, 0, stream>>>(trg_idx, wte, buf_x);
  for (int i = 0; i < 4; ++i) {
    ln_kernel<<<4096, 128, 0, stream>>>(buf_x, dec_ln1 + i * 512, buf_ln);
    gemm(buf_ln, dec_wqkv + (size_t)i * 512 * 1536, nullptr, ff1, 4096, 1536, 512, stream);
    attn_kernel<<<dim3(16, 8, 4), 128, 0, stream>>>(ff1, 1536, ff1 + 512, 1536, ff1 + 1024, 1536,
                                                    buf_attn, 512, tab_caus, 1024, 1024, 1);
    gemm(buf_attn, dec_wproj + (size_t)i * 512 * 512, buf_x, buf_x, 4096, 512, 512, stream);
    ln_kernel<<<4096, 128, 0, stream>>>(buf_x, dec_ln2 + i * 512, buf_ln);
    gemm(buf_ln, dec_wq + (size_t)i * 512 * 512, nullptr, ff2, 4096, 512, 512, stream);
    gemm(buf_enc, dec_wkv + (size_t)i * 512 * 1024, nullptr, buf_kv, 4096, 1024, 512, stream);
    attn_kernel<<<dim3(16, 8, 4), 128, 0, stream>>>(ff2, 512, buf_kv, 1024, buf_kv + 512, 1024,
                                                    buf_attn, 512, tab_bidir, 1024, 1024, 0);
    gemm(buf_attn, dec_wcproj + (size_t)i * 512 * 512, buf_x, buf_x, 4096, 512, 512, stream);
    ln_kernel<<<4096, 128, 0, stream>>>(buf_x, dec_ln3 + i * 512, buf_ln);
    gemm(buf_ln, dec_wgate + (size_t)i * 512 * 2048, nullptr, ff1, 4096, 2048, 512, stream);
    gemm(buf_ln, dec_wup + (size_t)i * 512 * 2048, nullptr, ff2, 4096, 2048, 512, stream);
    geglu_kernel<<<(8388608 + 255) / 256, 256, 0, stream>>>(ff1, ff2, ff1, 8388608);
    gemm(ff1, dec_wdown + (size_t)i * 2048 * 512, buf_x, buf_x, 4096, 512, 2048, stream);
  }
  ln_kernel<<<4096, 128, 0, stream>>>(buf_x, ln_dec, buf_ln);

  // tied lm head: logits = y @ wte^T  (N = 384 = 3*128)
  gemm(buf_ln, wteT, nullptr, out, 4096, 384, 512, stream);
}
